// LSTMPosUniversalModel_11802570129533
// MI455X (gfx1250) — matrix operand order repaired and checked by reading, not compiled, not run on hardware
//
#include <hip/hip_runtime.h>
#include <hip/hip_bf16.h>

// ---------------------------------------------------------------------------
// Types for CDNA5 WMMA (gfx1250, wave32)
// ---------------------------------------------------------------------------
typedef __bf16 bf16;
typedef __attribute__((ext_vector_type(16))) __bf16 v16bf;
typedef __attribute__((ext_vector_type(8)))  float  v8f;
typedef __attribute__((ext_vector_type(4)))  unsigned u32x4;
typedef __attribute__((ext_vector_type(8)))  int      i32x8;
typedef __attribute__((ext_vector_type(4)))  int      i32x4;

union TileAB { v16bf v; u32x4 q[2]; };

// Problem constants
#define Ee    512
#define Hh    1024
#define Ss    512
#define Bb    64
#define NTAGS 13
#define DFF   256
#define G4    4096          // 4*H
#define NWG   32            // workgroups in the scan grid
#define LH_LD 1032          // padded LDS row stride (elements) for h staging
#define LX_LD 520           // padded LDS row stride (elements) for x staging (TDM pad)

// ---------------------------------------------------------------------------
// WMMA operand loads per CDNA5 ISA VGPR striping (CDNA5 ISA §7.12.2)
// A 16x32 bf16: lane L row M=L%16; elems 0..7 -> K=k0+(L/16)*8+i,
//               elems 8..15 -> K=k0+16+(L/16)*8+(i-8)  (two 16B chunks 32B apart)
// B 32x16 bf16: lane L col N=L%16; same K striping as A (two 16B chunks 32B apart)
// ---------------------------------------------------------------------------
__device__ __forceinline__ v16bf load_a_tile(const bf16* base, int ld) {
  const int lane = threadIdx.x & 31;
  const bf16* p = base + (lane & 15) * ld + ((lane >> 4) << 3);
  TileAB t;
  t.q[0] = *(const u32x4*)(p);
  t.q[1] = *(const u32x4*)(p + 16);
  return t.v;
}

__device__ __forceinline__ v16bf load_b_tile(const bf16* base, int ld) {
  const int lane = threadIdx.x & 31;
  const bf16* p = base + (lane & 15) * ld + ((lane >> 4) << 3);
  TileAB t;
  t.q[0] = *(const u32x4*)(p);
  t.q[1] = *(const u32x4*)(p + 16);
  return t.v;
}

__device__ __forceinline__ v8f wmma_bf16(v16bf a, v16bf b, v8f c) {
  return __builtin_amdgcn_wmma_f32_16x16x32_bf16(false, a, false, b, (short)0, c,
                                                 false, false);
}

__device__ __forceinline__ float fast_sigmoid(float x) {
  return __builtin_amdgcn_rcpf(1.0f + __expf(-x));
}
__device__ __forceinline__ float fast_tanh(float x) {
  // 1 - 2/(e^{2x}+1): saturates to +/-1 correctly at large |x|
  return 1.0f - 2.0f * __builtin_amdgcn_rcpf(__expf(2.0f * x) + 1.0f);
}

// ---------------------------------------------------------------------------
// Grid-wide sense-reversing barrier (agent-scope atomics through L2)
// ---------------------------------------------------------------------------
__device__ __forceinline__ void grid_sync(unsigned* cnt, unsigned* gen) {
  __syncthreads();
  if (threadIdx.x == 0) {
    __threadfence();
    unsigned g = __hip_atomic_load(gen, __ATOMIC_RELAXED, __HIP_MEMORY_SCOPE_AGENT);
    unsigned a = __hip_atomic_fetch_add(cnt, 1u, __ATOMIC_ACQ_REL, __HIP_MEMORY_SCOPE_AGENT);
    if (a == NWG - 1u) {
      __hip_atomic_store(cnt, 0u, __ATOMIC_RELAXED, __HIP_MEMORY_SCOPE_AGENT);
      __hip_atomic_fetch_add(gen, 1u, __ATOMIC_RELEASE, __HIP_MEMORY_SCOPE_AGENT);
    } else {
      while (__hip_atomic_load(gen, __ATOMIC_ACQUIRE, __HIP_MEMORY_SCOPE_AGENT) == g) {
        __builtin_amdgcn_s_sleep(2);
      }
    }
  }
  __syncthreads();
}

// ---------------------------------------------------------------------------
// Prep kernels
// ---------------------------------------------------------------------------
__global__ void cvt_bf16_kernel(const float* __restrict__ src, bf16* __restrict__ dst,
                                long n) {
  long i = (long)blockIdx.x * blockDim.x + threadIdx.x;
  long stride = (long)gridDim.x * blockDim.x;
  for (; i < n; i += stride) dst[i] = (bf16)src[i];
}

__global__ void bias_sum_kernel(const float* __restrict__ a, const float* __restrict__ b,
                                float* __restrict__ o, int n) {
  int i = blockIdx.x * blockDim.x + threadIdx.x;
  if (i < n) o[i] = a[i] + b[i];
}

// x_bf[s][b][e] = bf16(emb[tokens[b][s]][e]); layout [S][B][E]
__global__ void gather_kernel(const int* __restrict__ tokens,
                              const float* __restrict__ emb,
                              bf16* __restrict__ xbf) {
  long n = (long)Ss * Bb * Ee;
  long i = (long)blockIdx.x * blockDim.x + threadIdx.x;
  long stride = (long)gridDim.x * blockDim.x;
  for (; i < n; i += stride) {
    int e = (int)(i & (Ee - 1));
    int b = (int)((i >> 9) & (Bb - 1));
    int s = (int)(i >> 15);
    long row = (long)tokens[b * Ss + s];
    xbf[i] = (bf16)emb[row * Ee + e];
  }
}

__global__ void init_barrier_kernel(unsigned* cnt, unsigned* gen) {
  if (threadIdx.x == 0) { *cnt = 0u; *gen = 0u; }
}

// ---------------------------------------------------------------------------
// Persistent LSTM scan kernel (32 WGs x 256 thr = 8 waves).
// WG owns hidden units [wg*32, wg*32+32); wave w: gate g=w>>1, n-tile t=w&1.
// Per step: gates = [x_t | h] @ [W_ih|W_hh](tag)^T (48 K-tiles), bf16 WMMA,
// bias pre-splatted into accumulators. x_t staged via TDM tensor_load_to_lds
// (D# pad fields produce the padded-row LDS layout), h staged via
// global_load_async_to_lds_b128. Copy-free modulo-scheduled pipeline with
// source-uniform phase loops (x-phase / peeled boundary / h-phase) so all
// in-loop addressing is affine. c stays in registers across all 512 steps.
// ---------------------------------------------------------------------------
__global__ __launch_bounds__(256) void
lstm_scan_kernel(const bf16* __restrict__ xbf,        // [S][64][512]
                 const bf16* __restrict__ Wih,        // [13][4096][512]
                 const bf16* __restrict__ Whh,        // [13][4096][1024]
                 const float* __restrict__ bias,      // [13][4096]
                 const int*  __restrict__ tags,       // [512]
                 bf16* __restrict__ hbuf0,            // [64][1024]
                 bf16* __restrict__ hbuf1,            // [64][1024]
                 unsigned* cnt, unsigned* gen) {
  __shared__ bf16  lh[64 * LH_LD];   // staged h, padded rows (~129 KB)
  __shared__ bf16  lx[64 * LX_LD];   // staged x_t, TDM-padded rows (~65 KB)
  __shared__ float lg[64 * 128];     // gate exchange tile (32 KB)

  const int tid  = threadIdx.x;
  const int wg   = blockIdx.x;
  const int wave = tid >> 5;
  const int lane = tid & 31;
  const int gte  = wave >> 1;
  const int tnt  = wave & 1;
  const int j0   = wg * 32;
  const int ncol = gte * Hh + j0 + tnt * 16;
  const int hi8  = (lane >> 4) << 3;

  const int jj    = tid & 31;     // elementwise mapping
  const int mrow0 = tid >> 5;

  float c[8];
#pragma unroll
  for (int p = 0; p < 8; ++p) c[p] = 0.0f;

#pragma unroll
  for (int p = 0; p < 8; ++p)
    hbuf0[(mrow0 + 8 * p) * Hh + j0 + jj] = (bf16)0.0f;

  grid_sync(cnt, gen);

  const unsigned lhbase = (unsigned)(unsigned long long)(const void*)&lh[0];
  const unsigned lxbase = (unsigned)(unsigned long long)(const void*)&lx[0];

  for (int s = 0; s < Ss; ++s) {
    const int tag = tags[s];
    const bf16* hsrc = (s & 1) ? hbuf1 : hbuf0;
    bf16*       hdst = (s & 1) ? hbuf0 : hbuf1;

    // ---- TDM: x_t (64x512 bf16, 64KB contiguous) -> lx, padded rows -------
    if (wave == 0) {
      unsigned long long ga =
          (unsigned long long)(const void*)(xbf + (size_t)s * (Bb * Ee));
      // D# group0: count=1, lds_addr, 57b global addr, type=2
      u32x4 g0 = { 1u, lxbase, (unsigned)(ga & 0xffffffffu),
                   (unsigned)((ga >> 32) & 0x01ffffffu) | (2u << 30) };
      // D# group1: data_size=4B, pad_enable, pad_interval=256DW, pad_amount=4DW
      // tensor_dim0 = tile_dim0 = 16384 DW, tensor_dim1 = tile_dim1 = 1
      i32x8 g1 = { (int)((2u << 16) | (1u << 20) | (7u << 22) | (3u << 25)),
                   (int)(16384u << 16),   // tensor_dim0[15:0] @ [63:48]
                   (int)(1u << 16),       // tensor_dim1[15:0] @ [95:80]
                   (int)(16384u << 16),   // tile_dim0 @ [127:112]
                   (int)1,                // tile_dim1 @ [143:128]
                   (int)16384,            // tensor_dim0_stride[31:0]
                   0, 0 };
      i32x4 g2 = { 0, 0, 0, 0 };
      i32x4 g3 = { 0, 0, 0, 0 };
      i32x8 g4 = { 0, 0, 0, 0, 0, 0, 0, 0 };
      __builtin_amdgcn_tensor_load_to_lds(g0, g1, g2, g3, g4, 0);
    }

    // ---- async copy: full h (64x1024 bf16 = 128KB) -> lh ------------------
    {
      const int row0 = tid >> 7;               // 2 rows per round
      const int colb = (tid & 127) * 16;       // byte offset within row
      for (int r = 0; r < 32; ++r) {
        const int rr = row0 + r * 2;
        unsigned ldso = lhbase + (unsigned)(rr * (LH_LD * 2) + colb);
        unsigned long long ga =
            (unsigned long long)((const char*)hsrc + rr * (Hh * 2) + colb);
        asm volatile("global_load_async_to_lds_b128 %0, %1, off"
                     :: "v"(ldso), "v"(ga) : "memory");
      }
      asm volatile("s_wait_asynccnt 0x0" ::: "memory");
    }
    if (wave == 0) __builtin_amdgcn_s_wait_tensorcnt(0);
    __syncthreads();

    // ---- GEMM over 48 K-tiles (16 x-tiles + 32 h-tiles), pipelined --------
    const bf16* Wi = Wih + (size_t)tag * (G4 * Ee) + (size_t)ncol * Ee;
    const bf16* Wh = Whh + (size_t)tag * (G4 * Hh) + (size_t)ncol * Hh;

    const float bval = bias[tag * G4 + ncol + (lane & 15)];
    v8f binit;
#pragma unroll
    for (int r = 0; r < 8; ++r) binit[r] = bval;
    v8f acc[4] = { binit, binit, binit, binit };

    // source-uniform tile loaders (affine addressing, no per-tile selects)
    auto loadBx = [&](int t) -> v16bf { return load_b_tile(Wi + (t << 5), Ee); };
    auto loadBh = [&](int t) -> v16bf { return load_b_tile(Wh + (t << 5), Hh); };
    auto loadAx = [&](int t, v16bf* dst) {
#pragma unroll
      for (int mt = 0; mt < 4; ++mt)
        dst[mt] = load_a_tile(&lx[mt * 16 * LX_LD + (t << 5)], LX_LD);
    };
    auto loadAh = [&](int t, v16bf* dst) {
#pragma unroll
      for (int mt = 0; mt < 4; ++mt)
        dst[mt] = load_a_tile(&lh[mt * 16 * LH_LD + (t << 5)], LH_LD);
    };
    auto comp = [&](const v16bf* aset, v16bf b) {
#pragma unroll
      for (int mt = 0; mt < 4; ++mt) acc[mt] = wmma_bf16(aset[mt], b, acc[mt]);
    };

    v16bf a0[4], a1[4];
    loadAx(0, a0);
    v16bf bA0 = loadBx(0), bA1 = loadBx(1);
    v16bf bB0 = loadBx(2), bB1 = loadBx(3);

    // X phase: tiles 0..11 (refills stay within x source)
    for (int t = 0; t < 12; t += 4) {
      loadAx(t + 1, a1);
      comp(a0, bA0);
      loadAx(t + 2, a0);
      comp(a1, bA1);
      bA0 = loadBx(t + 4);
      bA1 = loadBx(t + 5);
      loadAx(t + 3, a1);
      comp(a0, bB0);
      loadAx(t + 4, a0);
      comp(a1, bB1);
      bB0 = loadBx(t + 6);
      bB1 = loadBx(t + 7);
    }
    // Peeled boundary iteration: tiles 12..15, refills from h source
    {
      loadAx(13, a1);
      comp(a0, bA0);
      loadAx(14, a0);
      comp(a1, bA1);
      bA0 = loadBh(0);
      bA1 = loadBh(1);
      loadAx(15, a1);
      comp(a0, bB0);
      loadAh(0, a0);
      comp(a1, bB1);
      bB0 = loadBh(2);
      bB1 = loadBh(3);
    }
    // H phase: h-tiles 0..27
    for (int t = 0; t < 28; t += 4) {
      loadAh(t + 1, a1);
      comp(a0, bA0);
      loadAh(t + 2, a0);
      comp(a1, bA1);
      bA0 = loadBh(t + 4);
      bA1 = loadBh(t + 5);
      loadAh(t + 3, a1);
      comp(a0, bB0);
      loadAh(t + 4, a0);
      comp(a1, bB1);
      bB0 = loadBh(t + 6);
      bB1 = loadBh(t + 7);
    }
    // epilogue: h-tiles 28..31 (a0 holds Ah(28), B ring holds Bh(28..31))
    loadAh(29, a1);
    comp(a0, bA0);
    loadAh(30, a0);
    comp(a1, bA1);
    loadAh(31, a1);
    comp(a0, bB0);
    comp(a1, bB1);

    // ---- scatter accumulators to LDS gate tile [m][gte*32 + local_j] ------
#pragma unroll
    for (int mt = 0; mt < 4; ++mt) {
#pragma unroll
      for (int r = 0; r < 8; ++r) {
        lg[(mt * 16 + hi8 + r) * 128 + wave * 16 + (lane & 15)] = acc[mt][r];
      }
    }
    __syncthreads();

    // ---- elementwise LSTM cell (bias already in gates) --------------------
#pragma unroll
    for (int p = 0; p < 8; ++p) {
      const int m = mrow0 + 8 * p;
      float iv = lg[m * 128 +  0 + jj];
      float fv = lg[m * 128 + 32 + jj];
      float gv = lg[m * 128 + 64 + jj];
      float ov = lg[m * 128 + 96 + jj];
      c[p] = fast_sigmoid(fv) * c[p] + fast_sigmoid(iv) * fast_tanh(gv);
      float hv = fast_sigmoid(ov) * fast_tanh(c[p]);
      hdst[m * Hh + j0 + jj] = (bf16)hv;
    }

    grid_sync(cnt, gen);
  }
}

// ---------------------------------------------------------------------------
// FF head: out[b] = sigmoid(b2 + W2 @ relu(W1 @ h[b] + b1))
// ---------------------------------------------------------------------------
__global__ __launch_bounds__(256) void
head_kernel(const bf16* __restrict__ h, const float* __restrict__ W1,
            const float* __restrict__ b1, const float* __restrict__ W2,
            const float* __restrict__ b2, float* __restrict__ out) {
  __shared__ float hrow[Hh];
  __shared__ float hid[DFF];
  const int b = blockIdx.x, t = threadIdx.x;
  for (int k = t; k < Hh; k += 256) hrow[k] = (float)h[b * Hh + k];
  __syncthreads();
  float acc = b1[t];
  const float* w = W1 + t * Hh;
  for (int k = 0; k < Hh; ++k) acc += hrow[k] * w[k];
  hid[t] = fmaxf(acc, 0.0f) * W2[t];
  __syncthreads();
  for (int off = 128; off > 0; off >>= 1) {
    if (t < off) hid[t] += hid[t + off];
    __syncthreads();
  }
  if (t == 0) out[b] = 1.0f / (1.0f + __expf(-(hid[0] + b2[0])));
}

// ---------------------------------------------------------------------------
// Launch
// ---------------------------------------------------------------------------
extern "C" void kernel_launch(void* const* d_in, const int* in_sizes, int n_in,
                              void* d_out, int out_size, void* d_ws, size_t ws_size,
                              hipStream_t stream) {
  const int*   tokens = (const int*)d_in[0];
  const int*   tags   = (const int*)d_in[1];
  const float* emb    = (const float*)d_in[2];
  const float* W_ih   = (const float*)d_in[3];
  const float* W_hh   = (const float*)d_in[4];
  const float* b_ih   = (const float*)d_in[5];
  const float* b_hh   = (const float*)d_in[6];
  const float* W1     = (const float*)d_in[7];
  const float* b1     = (const float*)d_in[8];
  const float* W2     = (const float*)d_in[9];
  const float* b2     = (const float*)d_in[10];
  float* out = (float*)d_out;

  char* ws = (char*)d_ws;
  size_t off = 0;
  auto alloc = [&](size_t bytes) -> void* {
    void* p = ws + off;
    off += (bytes + 255) & ~(size_t)255;
    return p;
  };

  bf16*  Wih_bf = (bf16*)alloc((size_t)NTAGS * G4 * Ee * 2);   // 54.5 MB
  bf16*  Whh_bf = (bf16*)alloc((size_t)NTAGS * G4 * Hh * 2);   // 109 MB
  float* bias   = (float*)alloc((size_t)NTAGS * G4 * 4);       // 0.2 MB
  bf16*  xbf    = (bf16*)alloc((size_t)Ss * Bb * Ee * 2);      // 33.5 MB
  bf16*  hbuf0  = (bf16*)alloc((size_t)Bb * Hh * 2);
  bf16*  hbuf1  = (bf16*)alloc((size_t)Bb * Hh * 2);
  unsigned* bar = (unsigned*)alloc(512);
  unsigned* cnt = bar;
  unsigned* gen = bar + 32;

  cvt_bf16_kernel<<<4096, 256, 0, stream>>>(W_ih, Wih_bf, (long)NTAGS * G4 * Ee);
  cvt_bf16_kernel<<<8192, 256, 0, stream>>>(W_hh, Whh_bf, (long)NTAGS * G4 * Hh);
  bias_sum_kernel<<<(NTAGS * G4 + 255) / 256, 256, 0, stream>>>(b_ih, b_hh, bias,
                                                                NTAGS * G4);
  gather_kernel<<<8192, 256, 0, stream>>>(tokens, emb, xbf);
  init_barrier_kernel<<<1, 64, 0, stream>>>(cnt, gen);
  lstm_scan_kernel<<<NWG, 256, 0, stream>>>(xbf, Wih_bf, Whh_bf, bias, tags,
                                            hbuf0, hbuf1, cnt, gen);
  head_kernel<<<Bb, 256, 0, stream>>>(hbuf0, W1, b1, W2, b2, out);
}
